// LocalTransformer_89893665505283
// MI455X (gfx1250) — compile-verified, matrix operand order checked by reading, not run on hardware
//
// MI455X (gfx1250, CDNA5) local point transformer.
// Strategy:
//  - FPS: 1 workgroup/batch, dist[] in LDS (64KB), argmax via wave32 shuffles.
//  - Ball query: 1 wave32 per centroid, ballot/popcount ordered append.
//  - Gather -> residual stream X [4096 groups][64 tok][256] f32 in workspace.
//  - Encoder layer: ONE fused kernel, 1 workgroup (8 waves) per group.
//    All GEMMs via v_wmma_f32_16x16x32_f16 (f32 accum), operands staged in LDS
//    (h, Q, K, V^T, softmax P, FF intermediate). ~135KB LDS / workgroup.
//  - Max-pool + final FC GEMM also via WMMA.
// Workspace requirement: ~280 MB (X=268MB dominates).
#include <hip/hip_runtime.h>
#include <hip/hip_bf16.h>

typedef _Float16 h16;
typedef __attribute__((ext_vector_type(16))) _Float16 v16h;
typedef __attribute__((ext_vector_type(8)))  _Float16 v8h;
typedef __attribute__((ext_vector_type(8)))  float    v8f;

#define BQ 4
#define NPTS 16384
#define NPOINT 1024
#define NS 64
#define DIM 256
#define FFD 512
#define GRP (BQ*NPOINT)      // 4096 groups
#define SH 264               // LDS f16 stride for 256-col buffers (16B aligned rows)
#define SVT 72               // LDS stride for V^T [256][72]
#define R2 1.44f             // radius^2

// ---------------- WMMA helpers (CDNA5 f16 16x16x32, f32 accum) ----------------
__device__ __forceinline__ v8f wmma16(v16h a, v16h b, v8f c) {
  return __builtin_amdgcn_wmma_f32_16x16x32_f16(false, a, false, b, (short)0, c, false, false);
}

// A-layout (and B-as-B^T) fragment pointer: lane<16 -> row=lane, K base 0;
// lane>=16 -> row=lane-16, K base 8. Halves 0..7 = K kb..kb+7, 8..15 = K kb+16..kb+23.
__device__ __forceinline__ const h16* tptr(const h16* base, int ld, int row0, int col0, int lane) {
  int r  = lane & 15;
  int kb = (lane & 16) ? 8 : 0;
  return base + (size_t)(row0 + r) * ld + col0 + kb;
}
__device__ __forceinline__ v16h ld_tile(const h16* p) {
  v8h lo = *(const v8h*)(p);
  v8h hi = *(const v8h*)(p + 16);
  return __builtin_shufflevector(lo, hi, 0,1,2,3,4,5,6,7,8,9,10,11,12,13,14,15);
}

__device__ __forceinline__ float wred32(float v) {
#pragma unroll
  for (int i = 16; i > 0; i >>= 1) v += __shfl_xor(v, i, 32);
  return v;
}

// ---------------- FPS ----------------
__global__ __launch_bounds__(1024) void fps_kernel(const float* __restrict__ xyz,
                                                   float* __restrict__ newxyz_out,
                                                   float* __restrict__ fpsidx_out) {
  __shared__ float dist[NPTS];
  __shared__ float rv[32];
  __shared__ int   ri[32];
  __shared__ int   s_cur;
  const int b = blockIdx.x, t = threadIdx.x, lane = t & 31, w = t >> 5;
  const float* xb = xyz + (size_t)b * NPTS * 3;
  float px[16], py[16], pz[16];
#pragma unroll
  for (int i = 0; i < 16; ++i) {
    int p = t + i * 1024;
    px[i] = xb[p*3]; py[i] = xb[p*3+1]; pz[i] = xb[p*3+2];
    dist[p] = 1e10f;
  }
  if (t == 0) s_cur = 0;
  __syncthreads();
  for (int it = 0; it < NPOINT; ++it) {
    int cur = s_cur;
    float cx = xb[cur*3], cy = xb[cur*3+1], cz = xb[cur*3+2];
    if (t == 0) {
      fpsidx_out[b*NPOINT + it] = (float)cur;          // int idx emitted as float
      newxyz_out[(b*NPOINT + it)*3 + 0] = cx;
      newxyz_out[(b*NPOINT + it)*3 + 1] = cy;
      newxyz_out[(b*NPOINT + it)*3 + 2] = cz;
    }
    float best = -1.f; int bidx = 0;
#pragma unroll
    for (int i = 0; i < 16; ++i) {
      int p = t + i * 1024;
      float dx = px[i]-cx, dy = py[i]-cy, dz = pz[i]-cz;
      float d  = dx*dx + dy*dy + dz*dz;
      float nd = fminf(dist[p], d);
      dist[p] = nd;
      if (nd > best) { best = nd; bidx = p; }
    }
#pragma unroll
    for (int i = 16; i > 0; i >>= 1) {
      float ov = __shfl_xor(best, i, 32);
      int   oi = __shfl_xor(bidx, i, 32);
      if (ov > best || (ov == best && oi < bidx)) { best = ov; bidx = oi; }
    }
    if (lane == 0) { rv[w] = best; ri[w] = bidx; }
    __syncthreads();
    if (w == 0) {
      float v2 = rv[lane]; int i2 = ri[lane];
#pragma unroll
      for (int i = 16; i > 0; i >>= 1) {
        float ov = __shfl_xor(v2, i, 32);
        int   oi = __shfl_xor(i2, i, 32);
        if (ov > v2 || (ov == v2 && oi < i2)) { v2 = ov; i2 = oi; }
      }
      if (lane == 0) s_cur = i2;
    }
    __syncthreads();
  }
}

// ---------------- Ball query: one wave per centroid ----------------
__global__ __launch_bounds__(256) void ball_kernel(const float* __restrict__ xyz,
                                                   const float* __restrict__ newxyz,
                                                   int* __restrict__ idxbuf) {
  const int lane = threadIdx.x & 31;
  const int c = blockIdx.x * 8 + (threadIdx.x >> 5);
  const int b = c >> 10;
  const float* xb = xyz + (size_t)b * NPTS * 3;
  const float cx = newxyz[c*3], cy = newxyz[c*3+1], cz = newxyz[c*3+2];
  int* out = idxbuf + (size_t)c * NS;
  int cnt = 0, first = 0;
  for (int base = 0; base < NPTS && cnt < NS; base += 32) {
    int p = base + lane;
    float dx = xb[p*3]-cx, dy = xb[p*3+1]-cy, dz = xb[p*3+2]-cz;
    bool hit = (dx*dx + dy*dy + dz*dz) < R2;
    unsigned mm = (unsigned)__ballot(hit);
    if (cnt == 0 && mm) first = base + __ffs(mm) - 1;
    int rank = __popc(mm & ((1u << lane) - 1u));
    if (hit && (cnt + rank) < NS) out[cnt + rank] = p;
    cnt += __popc(mm);
  }
  if (cnt == 0)      { for (int j = lane;       j < NS; j += 32) out[j] = 0;     }
  else if (cnt < NS) { for (int j = cnt + lane; j < NS; j += 32) out[j] = first; }
}

// ---------------- Gather grouped features + relative xyz ----------------
__global__ __launch_bounds__(256) void gather_kernel(const float* __restrict__ feat,
                                                     const float* __restrict__ xyz,
                                                     const float* __restrict__ newxyz,
                                                     const int* __restrict__ idxbuf,
                                                     float* __restrict__ X,
                                                     float* __restrict__ gxyz) {
  const int g = blockIdx.x, t = threadIdx.x, b = g >> 10;
  const int* idl = idxbuf + (size_t)g * NS;
  float* xo = X + (size_t)g * NS * DIM;
  const float* fb = feat + (size_t)b * DIM * NPTS;
  for (int s = 0; s < NS; ++s) {
    int i = idl[s];
    xo[s * DIM + t] = fb[(size_t)t * NPTS + i];
  }
  if (t < NS * 3) {
    int s = t / 3, k = t % 3;
    int i = idl[s];
    gxyz[(size_t)g * NS * 3 + t] = xyz[((size_t)b * NPTS + i) * 3 + k] - newxyz[g * 3 + k];
  }
}

// ---------------- f32 -> f16 weight conversion ----------------
__global__ void f2h_kernel(const float* __restrict__ s, h16* __restrict__ d, int n) {
  int i = blockIdx.x * 256 + threadIdx.x;
  if (i < n) d[i] = (h16)s[i];
}

// ---------------- LayerNorm into f16 LDS buffer (one wave handles 8 tokens) ----
__device__ __forceinline__ void layernorm_to(const float* xg, h16* dst,
                                             const float* gw, const float* gb,
                                             int wv_id, int lane) {
  for (int t = wv_id * 8; t < wv_id * 8 + 8; ++t) {
    const float* xr = xg + t * DIM;
    float vals[8]; float s = 0.f;
#pragma unroll
    for (int i = 0; i < 8; ++i) { vals[i] = xr[lane * 8 + i]; s += vals[i]; }
    s = wred32(s);
    float mean = s * (1.f / DIM);
    float vs = 0.f;
#pragma unroll
    for (int i = 0; i < 8; ++i) { float d = vals[i] - mean; vs += d * d; }
    vs = wred32(vs);
    float rstd = rsqrtf(vs * (1.f / DIM) + 1e-5f);
#pragma unroll
    for (int i = 0; i < 8; ++i) {
      int c = lane * 8 + i;
      dst[t * SH + c] = (h16)((vals[i] - mean) * rstd * gw[c] + gb[c]);
    }
  }
}

// ---------------- Fused encoder layer: 1 workgroup (8 waves) per group --------
__global__ __launch_bounds__(256) void layer_kernel(
    float* __restrict__ X, const float* __restrict__ gxyz,
    const h16* __restrict__ wq, const h16* __restrict__ wk, const h16* __restrict__ wv,
    const h16* __restrict__ wo, const h16* __restrict__ w1, const h16* __restrict__ w2,
    const float* __restrict__ bq, const float* __restrict__ bk, const float* __restrict__ bvv,
    const float* __restrict__ bo, const float* __restrict__ b1, const float* __restrict__ b2,
    const float* __restrict__ ln1g, const float* __restrict__ ln1b,
    const float* __restrict__ ln2g, const float* __restrict__ ln2b,
    const float* __restrict__ pew, const float* __restrict__ peb) {
  __shared__ h16 sA[64 * SH];     // h (LN1), then softmax P, then h2 (LN2)
  __shared__ h16 sQ[64 * SH];     // Q, then attention output O
  __shared__ h16 sK[64 * SH];     // K, then FF intermediate u[:, 0:256)
  __shared__ h16 sVt[256 * SVT];  // V^T [dh=256][tok=64], then u[:, 256:512)
  __shared__ float sS[64];        // s_t = pe_w . xyz_t  (rank-1 attention bias)

  const int g = blockIdx.x;
  const int tid = threadIdx.x;
  const int wv_id = tid >> 5;
  const int lane = tid & 31;
  const int n_lane = lane & 15;
  const int mb = (lane & 16) ? 8 : 0;
  float* xg = X + (size_t)g * NS * DIM;

  // Warm shared weights toward the caches (global_prefetch_b8); all 4096
  // workgroups stream the same ~1MB of f16 weights, L2-resident after warmup.
  __builtin_prefetch(wq, 0, 1);
  __builtin_prefetch(wk, 0, 1);
  __builtin_prefetch(wv, 0, 1);
  __builtin_prefetch(wo, 0, 1);
  __builtin_prefetch(w1, 0, 1);
  __builtin_prefetch(w2, 0, 1);

  // Stage 1: LN1 -> sA, plus rank-1 bias scalars
  layernorm_to(xg, sA, ln1g, ln1b, wv_id, lane);
  if (tid < 64) {
    const float* p = gxyz + ((size_t)g * NS + tid) * 3;
    sS[tid] = pew[0] * p[0] + pew[1] * p[1] + pew[2] * p[2];
  }
  __syncthreads();

  // Stage 2: Q,K,V = LN1(x) @ W^T + b ; V stored transposed for the P@V GEMM
  for (int j = wv_id; j < 192; j += 8) {
    int which = j >> 6;            // 0=Q 1=K 2=V
    int t = j & 63, mt = t & 3, nt = t >> 2;
    const h16*  W = which == 0 ? wq : which == 1 ? wk : wv;
    const float* bias = which == 0 ? bq : which == 1 ? bk : bvv;
    v8f acc = {0.f,0.f,0.f,0.f,0.f,0.f,0.f,0.f};
    for (int k0 = 0; k0 < DIM; k0 += 32) {
      v16h a = ld_tile(tptr(sA, SH, mt * 16, k0, lane));
      v16h b = ld_tile(tptr(W, DIM, nt * 16, k0, lane));
      acc = wmma16(a, b, acc);
    }
    int col = nt * 16 + n_lane;
    float bb = bias[col];
    if (which == 2) {
#pragma unroll
      for (int r = 0; r < 8; ++r) sVt[col * SVT + mt * 16 + mb + r] = (h16)(acc[r] + bb);
    } else {
      h16* dst = which == 0 ? sQ : sK;
#pragma unroll
      for (int r = 0; r < 8; ++r) dst[(mt * 16 + mb + r) * SH + col] = (h16)(acc[r] + bb);
    }
  }
  __syncthreads();

  // Stage 3: attention; 16 jobs = (head, qtile), each wave takes 2
  const float peb0 = peb[0];
  for (int j = wv_id; j < 16; j += 8) {
    int hh = j >> 2, qt = j & 3;
    v8f sc[4];
    for (int kt = 0; kt < 4; ++kt) {
      v8f acc = {0.f,0.f,0.f,0.f,0.f,0.f,0.f,0.f};
#pragma unroll
      for (int ks = 0; ks < 2; ++ks) {
        v16h a = ld_tile(tptr(sQ, SH, qt * 16, hh * 64 + ks * 32, lane));
        v16h b = ld_tile(tptr(sK, SH, kt * 16, hh * 64 + ks * 32, lane));
        acc = wmma16(a, b, acc);
      }
      sc[kt] = acc;
    }
#pragma unroll
    for (int kt = 0; kt < 4; ++kt)
#pragma unroll
      for (int r = 0; r < 8; ++r) {
        int q = qt * 16 + mb + r, k = kt * 16 + n_lane;
        sc[kt][r] = sc[kt][r] * 0.125f + sS[q] - sS[k] + peb0;
      }
    // softmax over 64 keys: per row reduce over 4 kt regs + 16-lane half shuffles
#pragma unroll
    for (int r = 0; r < 8; ++r) {
      float m = fmaxf(fmaxf(sc[0][r], sc[1][r]), fmaxf(sc[2][r], sc[3][r]));
#pragma unroll
      for (int i = 8; i > 0; i >>= 1) m = fmaxf(m, __shfl_xor(m, i, 32));
      float e[4]; float s = 0.f;
#pragma unroll
      for (int kt = 0; kt < 4; ++kt) { e[kt] = __expf(sc[kt][r] - m); s += e[kt]; }
#pragma unroll
      for (int i = 8; i > 0; i >>= 1) s += __shfl_xor(s, i, 32);
      float inv = 1.f / s;
      int q = qt * 16 + mb + r;
#pragma unroll
      for (int kt = 0; kt < 4; ++kt)
        sA[q * SH + hh * 64 + kt * 16 + n_lane] = (h16)(e[kt] * inv);
    }
    // O = P @ V_head  (B from transposed V, row-major)
    for (int nt = 0; nt < 4; ++nt) {
      v8f acc = {0.f,0.f,0.f,0.f,0.f,0.f,0.f,0.f};
#pragma unroll
      for (int ks = 0; ks < 2; ++ks) {
        v16h a = ld_tile(tptr(sA, SH, qt * 16, hh * 64 + ks * 32, lane));
        v16h b = ld_tile(tptr(sVt, SVT, hh * 64 + nt * 16, ks * 32, lane));
        acc = wmma16(a, b, acc);
      }
#pragma unroll
      for (int r = 0; r < 8; ++r)
        sQ[(qt * 16 + mb + r) * SH + hh * 64 + nt * 16 + n_lane] = (h16)acc[r];
    }
  }
  __syncthreads();

  // Stage 4: x1 = x + O @ Wo^T + bo  (in-place in global residual stream)
  for (int t = wv_id; t < 64; t += 8) {
    int mt = t & 3, nt = t >> 2;
    v8f acc = {0.f,0.f,0.f,0.f,0.f,0.f,0.f,0.f};
    for (int k0 = 0; k0 < DIM; k0 += 32) {
      v16h a = ld_tile(tptr(sQ, SH, mt * 16, k0, lane));
      v16h b = ld_tile(tptr(wo, DIM, nt * 16, k0, lane));
      acc = wmma16(a, b, acc);
    }
    int col = nt * 16 + n_lane;
    float bb = bo[col];
#pragma unroll
    for (int r = 0; r < 8; ++r) {
      float* px = xg + (mt * 16 + mb + r) * DIM + col;
      *px = *px + acc[r] + bb;
    }
  }
  __threadfence_block();
  __syncthreads();

  // Stage 5: LN2 -> sA
  layernorm_to(xg, sA, ln2g, ln2b, wv_id, lane);
  __syncthreads();

  // Stage 6: u = relu(h2 @ W1^T + b1), u[64][512] split across sK / sVt regions
  for (int j = wv_id; j < 128; j += 8) {
    int mt = j & 3, nt = j >> 2;   // nt in [0,32)
    v8f acc = {0.f,0.f,0.f,0.f,0.f,0.f,0.f,0.f};
    for (int k0 = 0; k0 < DIM; k0 += 32) {
      v16h a = ld_tile(tptr(sA, SH, mt * 16, k0, lane));
      v16h b = ld_tile(tptr(w1, DIM, nt * 16, k0, lane));
      acc = wmma16(a, b, acc);
    }
    int col = nt * 16 + n_lane;
    float bb = b1[col];
#pragma unroll
    for (int r = 0; r < 8; ++r) {
      float v = fmaxf(acc[r] + bb, 0.f);
      int row = mt * 16 + mb + r;
      if (col < 256) sK[row * SH + col] = (h16)v;
      else           sVt[row * SH + (col - 256)] = (h16)v;
    }
  }
  __syncthreads();

  // Stage 7: x2 = x1 + u @ W2^T + b2
  for (int t = wv_id; t < 64; t += 8) {
    int mt = t & 3, nt = t >> 2;
    v8f acc = {0.f,0.f,0.f,0.f,0.f,0.f,0.f,0.f};
    for (int kt = 0; kt < 16; ++kt) {
      int k0 = kt * 32;
      const h16* ubase = (k0 < 256) ? sK : sVt;
      int c0 = (k0 < 256) ? k0 : k0 - 256;
      v16h a = ld_tile(tptr(ubase, SH, mt * 16, c0, lane));
      v16h b = ld_tile(tptr(w2, FFD, nt * 16, k0, lane));
      acc = wmma16(a, b, acc);
    }
    int col = nt * 16 + n_lane;
    float bb = b2[col];
#pragma unroll
    for (int r = 0; r < 8; ++r) {
      float* px = xg + (mt * 16 + mb + r) * DIM + col;
      *px = *px + acc[r] + bb;
    }
  }
}

// ---------------- Max-pool over samples ----------------
__global__ __launch_bounds__(256) void pool_kernel(const float* __restrict__ X,
                                                   h16* __restrict__ pooled16) {
  const int g = blockIdx.x, d = threadIdx.x;
  const float* xg = X + (size_t)g * NS * DIM;
  float m = -3.402823466e38f;
  for (int s = 0; s < NS; ++s) m = fmaxf(m, xg[s * DIM + d]);
  pooled16[(size_t)g * DIM + d] = (h16)m;
}

// ---------------- Final FC: out[b,o,n] = fcw[o,:] . pooled[b,n,:] + fcb[o] ----
__global__ __launch_bounds__(256) void fc_kernel(const h16* __restrict__ pooled16,
                                                 const h16* __restrict__ fcw16,
                                                 const float* __restrict__ fcb,
                                                 float* __restrict__ out) {
  const int wglob = blockIdx.x * 8 + (threadIdx.x >> 5);   // 4096 tile jobs
  const int lane = threadIdx.x & 31;
  const int mt = wglob >> 4, nt = wglob & 15;
  __builtin_prefetch(fcw16, 0, 1);
  v8f acc = {0.f,0.f,0.f,0.f,0.f,0.f,0.f,0.f};
  for (int k0 = 0; k0 < DIM; k0 += 32) {
    v16h a = ld_tile(tptr(pooled16, DIM, mt * 16, k0, lane));
    v16h b = ld_tile(tptr(fcw16, DIM, nt * 16, k0, lane));
    acc = wmma16(a, b, acc);
  }
  const int n_lane = lane & 15, mb = (lane & 16) ? 8 : 0;
  const int o = nt * 16 + n_lane;
  const float bb = fcb[o];
#pragma unroll
  for (int r = 0; r < 8; ++r) {
    int row = mt * 16 + mb + r;          // row = b*1024 + n
    int b = row >> 10, np = row & 1023;
    out[((size_t)b * DIM + o) * NPOINT + np] = acc[r] + bb;
  }
}

// ---------------- Host orchestration ----------------
extern "C" void kernel_launch(void* const* d_in, const int* in_sizes, int n_in,
                              void* d_out, int out_size, void* d_ws, size_t ws_size,
                              hipStream_t stream) {
  (void)in_sizes; (void)n_in; (void)out_size; (void)ws_size;
  // JAX tree-flatten order: xyz, features, fc.b, fc.w, then per layer (sorted):
  // Wk.b, Wk.w, Wo.b, Wo.w, Wq.b, Wq.w, Wv.b, Wv.w, ff1.b, ff1.w, ff2.b, ff2.w,
  // ln1_b, ln1_g, ln2_b, ln2_g, pe.b, pe.w
  const float* xyz  = (const float*)d_in[0];
  const float* feat = (const float*)d_in[1];
  const float* fcb  = (const float*)d_in[2];
  const float* fcw  = (const float*)d_in[3];
  auto LP = [&](int layer, int k) -> const float* {
    return (const float*)d_in[4 + layer * 18 + k];
  };

  char* ws = (char*)d_ws;
  float* X        = (float*)ws;              size_t off = (size_t)GRP * NS * DIM * 4;
  float* gxyz     = (float*)(ws + off);      off += (size_t)GRP * NS * 3 * 4;
  int*   idxbuf   = (int*)(ws + off);        off += (size_t)GRP * NS * 4;
  h16*   pooled16 = (h16*)(ws + off);        off += (size_t)GRP * DIM * 2;
  h16*   w16      = (h16*)(ws + off);
  auto W16 = [&](int layer, int which) -> h16* {
    const size_t offs[6] = {0, 65536, 131072, 196608, 262144, 393216}; // q,k,v,o,w1,w2
    return w16 + (size_t)layer * 524288 + offs[which];
  };
  h16* fcw16 = w16 + 2 * 524288;

  float* out_newxyz = (float*)d_out;
  float* out_feat   = out_newxyz + (size_t)BQ * NPOINT * 3;
  float* out_fps    = out_feat + (size_t)BQ * DIM * NPOINT;

  fps_kernel<<<BQ, 1024, 0, stream>>>(xyz, out_newxyz, out_fps);
  ball_kernel<<<GRP / 8, 256, 0, stream>>>(xyz, out_newxyz, idxbuf);
  gather_kernel<<<GRP, 256, 0, stream>>>(feat, xyz, out_newxyz, idxbuf, X, gxyz);

  for (int l = 0; l < 2; ++l) {
    f2h_kernel<<<65536 / 256, 256, 0, stream>>>(LP(l, 5),  W16(l, 0), 65536);   // Wq.w
    f2h_kernel<<<65536 / 256, 256, 0, stream>>>(LP(l, 1),  W16(l, 1), 65536);   // Wk.w
    f2h_kernel<<<65536 / 256, 256, 0, stream>>>(LP(l, 7),  W16(l, 2), 65536);   // Wv.w
    f2h_kernel<<<65536 / 256, 256, 0, stream>>>(LP(l, 3),  W16(l, 3), 65536);   // Wo.w
    f2h_kernel<<<131072 / 256, 256, 0, stream>>>(LP(l, 9),  W16(l, 4), 131072); // ff1.w
    f2h_kernel<<<131072 / 256, 256, 0, stream>>>(LP(l, 11), W16(l, 5), 131072); // ff2.w
  }
  f2h_kernel<<<65536 / 256, 256, 0, stream>>>(fcw, fcw16, 65536);

  for (int l = 0; l < 2; ++l) {
    layer_kernel<<<GRP, 256, 0, stream>>>(
        X, gxyz,
        W16(l, 0), W16(l, 1), W16(l, 2), W16(l, 3), W16(l, 4), W16(l, 5),
        LP(l, 4), LP(l, 0), LP(l, 6), LP(l, 2), LP(l, 8), LP(l, 10),  // bq,bk,bv,bo,b1,b2
        LP(l, 13), LP(l, 12), LP(l, 15), LP(l, 14),                   // ln1g,ln1b,ln2g,ln2b
        LP(l, 17), LP(l, 16));                                        // pe.w, pe.b
  }

  pool_kernel<<<GRP, 256, 0, stream>>>(X, pooled16);
  // (GRP/16) M-tiles x 16 N-tiles = 4096 wave-jobs, 8 waves per 256-thread block
  fc_kernel<<<(GRP / 16) * 16 / 8, 256, 0, stream>>>(pooled16, fcw16, fcb, out_feat);
}
